// CLSAEncoder_86328842650104
// MI455X (gfx1250) — matrix operand order repaired and checked by reading, not compiled, not run on hardware
//
#include <hip/hip_runtime.h>

// CLSA encoder for MI455X (gfx1250, wave32).
// 8 persistent workgroups (one per batch lane — the only coarse parallelism in
// the sequential scan). h-states in LDS; c-states + 128MB attention history in
// d_ws (fully L2-resident on a 192MB-L2 part). Conv gates = bf16 WMMA GEMMs
// with a k-major K-ordering so B-fragments are 2x ds_load_b128 each.
// Attention matvecs are float4 VALU streams. Next x-frame staged by TDM.

#define ROWSN 32
#define COLSN 32
#define CHN   8
#define TN    128
#define BN    8
#define DN    (ROWSN * COLSN * CHN)   // 8192

#define NTHREADS 512
#define NWAVES   (NTHREADS / 32)
#define NTILES   ((ROWSN * COLSN) / 16)   // 64 N-tiles per batch

typedef __attribute__((ext_vector_type(16))) __bf16 v16bf;
typedef __attribute__((ext_vector_type(8)))  float  v8f;
typedef __attribute__((ext_vector_type(4)))  unsigned u32x4;
typedef __attribute__((ext_vector_type(8)))  int      i32x8;
typedef __attribute__((ext_vector_type(4)))  int      i32x4;

union AFrag { v16bf v; __bf16 e[16]; uint4 u[2]; };
union CFrag { v8f   v; float  e[8];  };

struct __align__(32) SMem {
  __bf16 zbp[ROWSN][COLSN + 2][16];   // [row][col+halo][ic] patches, 34KB
  float  h0[DN];                      // 32KB
  float  h1[DN];                      // 32KB
  float  xstage[ROWSN * COLSN];       // TDM-staged x frame, 4KB
  float  sc[TN];
  float  wb[TN];
  float  bias0[32];
  float  bias1[32];
};

__device__ __forceinline__ __bf16 tobf(float f) {
  unsigned u = __float_as_uint(f);
  u += 0x7FFFu + ((u >> 16) & 1u);    // RNE
  return __builtin_bit_cast(__bf16, (unsigned short)(u >> 16));
}

__device__ __forceinline__ float fastrcp(float x) {
#if __has_builtin(__builtin_amdgcn_rcpf)
  return __builtin_amdgcn_rcpf(x);
#else
  return 1.0f / x;
#endif
}
__device__ __forceinline__ float sigf(float x) {     // branch-free sigmoid
  return fastrcp(1.0f + __expf(-x));
}
__device__ __forceinline__ float tanhf_fast(float x) { // branch-free tanh
  return 1.0f - 2.0f * fastrcp(1.0f + __expf(2.0f * x));
}

#if __has_builtin(__builtin_amdgcn_tensor_load_to_lds)
#define HAVE_TDM 1
__device__ __forceinline__ void tdm_load_frame(const float* gsrc, unsigned lds_off) {
  unsigned long long ga = (unsigned long long)(uintptr_t)gsrc;
  u32x4 g0;
  g0[0] = 1u;                                     // count=1 user descriptor
  g0[1] = lds_off;                                // lds_addr (bytes)
  g0[2] = (unsigned)(ga & 0xFFFFFFFFu);           // global_addr[31:0]
  g0[3] = (unsigned)((ga >> 32) & 0x01FFFFFFu) | (2u << 30);  // [56:32] | type=2
  i32x8 g1;
  g1[0] = (int)(2u << 16);                        // data_size = 4B
  g1[1] = (int)(0x0400u << 16);                   // tensor_dim0 = 1024
  g1[2] = (int)(1u << 16);                        // tensor_dim1 = 1
  g1[3] = (int)(0x0400u << 16);                   // tile_dim0 = 1024
  g1[4] = 1;                                      // tile_dim1 = 1
  g1[5] = 1024;                                   // tensor_dim0_stride
  g1[6] = 0; g1[7] = 0;
  i32x4 gz = {0, 0, 0, 0};
#if defined(__clang_major__) && __clang_major__ >= 23
  i32x8 gz8 = {0, 0, 0, 0, 0, 0, 0, 0};
  __builtin_amdgcn_tensor_load_to_lds(g0, g1, gz, gz, gz8, 0);
#else
  __builtin_amdgcn_tensor_load_to_lds(g0, g1, gz, gz, 0);
#endif
}
#else
#define HAVE_TDM 0
#endif

// Conv-as-GEMM + LSTM gates. K-order kappa' = k*16+ic; chunk q covers
// k = {2q (lanes 0-15), 2q+1 (lanes 16-31)}; k==3 half is the zero pad.
__device__ __forceinline__ void conv_cell(SMem& sm, const AFrag* amt0,
                                          const AFrag* amt1,
                                          const float* biasLds, float* cglob,
                                          float* hLds, int wave, int lane) {
  const bool hi   = lane >= 16;
  const int  nloc = lane & 15;
  float blo[8], bhi[8];
#pragma unroll
  for (int r = 0; r < 8; ++r) {
    blo[r] = biasLds[(hi ? 8 : 0) + r];
    bhi[r] = biasLds[16 + (hi ? 8 : 0) + r];
  }
  for (int i = 0; i < NTILES / NWAVES; ++i) {
    const int nt  = wave + NWAVES * i;
    const int row = nt >> 1;
    const int col = (nt & 1) * 16 + nloc;
    CFrag d0, d1;
#pragma unroll
    for (int r = 0; r < 8; ++r) { d0.e[r] = 0.f; d1.e[r] = 0.f; }
#pragma unroll
    for (int q = 0; q < 2; ++q) {
      const int kc = 2 * q + (hi ? 1 : 0);        // fragment-half k value
      AFrag bf;
      bf.v = (v16bf)tobf(0.0f);
      if (kc < 3) {                                // k==3 -> zero pad half
        const uint4* p4 =
            reinterpret_cast<const uint4*>(&sm.zbp[row][col + kc][0]);
        bf.u[0] = p4[0];                           // 2x ds_load_b128
        bf.u[1] = p4[1];
      }
      d0.v = __builtin_amdgcn_wmma_f32_16x16x32_bf16(
          false, amt0[q].v, false, bf.v, (short)0, d0.v, false, false);
      d1.v = __builtin_amdgcn_wmma_f32_16x16x32_bf16(
          false, amt1[q].v, false, bf.v, (short)0, d1.v, false, false);
    }
#pragma unroll
    for (int r = 0; r < 8; ++r) {
      const float g0 = d0.e[r] + blo[r];          // i (lo) | f (hi)
      const float g1 = d1.e[r] + bhi[r];          // g (lo) | o (hi)
      const float p0 = __shfl_xor(g0, 16, 32);
      const float p1 = __shfl_xor(g1, 16, 32);
      const float iv = hi ? p0 : g0;
      const float fv = hi ? g0 : p0;
      const float gv = hi ? p1 : g1;
      const float ov = hi ? g1 : p1;
      const int idx  = (row * CHN + r) * COLSN + col;
      const float cn = sigf(fv) * cglob[idx] + sigf(iv) * tanhf_fast(gv);
      const float hn = sigf(ov) * tanhf_fast(cn);
      if (!hi) { cglob[idx] = cn; hLds[idx] = hn; }
    }
  }
}

// Per-batch attention: matvec scores + masked softmax + ctx AXPY (float4
// streams over the L2-resident history), history append.
__device__ __forceinline__ void attention(SMem& sm, const float* qv,
                                          float* rawb, float* refb,
                                          float* attn_row, int s,
                                          float* r_out_global, bool r_to_zb,
                                          int tid, int wave, int lane) {
  const float4* qp = reinterpret_cast<const float4*>(qv);
  if (s > 0) {
    for (int t2 = wave; t2 < s; t2 += NWAVES) {
      const float4* rp = reinterpret_cast<const float4*>(rawb + (size_t)t2 * DN);
      float p = 0.f;
      for (int d = lane; d < DN / 4; d += 32) {
        const float4 a = qp[d], b = rp[d];
        p = fmaf(a.x, b.x, p); p = fmaf(a.y, b.y, p);
        p = fmaf(a.z, b.z, p); p = fmaf(a.w, b.w, p);
      }
#pragma unroll
      for (int o = 16; o; o >>= 1) p += __shfl_xor(p, o, 32);
      if (lane == 0) sm.sc[t2] = p * 0.01104854345603993f;   // 1/sqrt(8192)
    }
    __syncthreads();
    if (wave == 0) {
      float m = -3.0e38f;
      for (int t2 = lane; t2 < s; t2 += 32) m = fmaxf(m, sm.sc[t2]);
#pragma unroll
      for (int o = 16; o; o >>= 1) m = fmaxf(m, __shfl_xor(m, o, 32));
      float sum = 0.f;
      for (int t2 = lane; t2 < s; t2 += 32) {
        const float e = __expf(sm.sc[t2] - m);
        sm.wb[t2] = e; sum += e;
      }
#pragma unroll
      for (int o = 16; o; o >>= 1) sum += __shfl_xor(sum, o, 32);
      const float inv = fastrcp(sum);
      for (int t2 = lane; t2 < s; t2 += 32) sm.wb[t2] *= inv;
    }
    __syncthreads();
  }
  for (int t2 = tid; t2 < TN; t2 += NTHREADS)
    attn_row[t2] = (t2 < s) ? sm.wb[t2] : 0.f;
  for (int d4 = tid; d4 < DN / 4; d4 += NTHREADS) {
    const float4 q4 = qp[d4];
    float4 acc = {0.f, 0.f, 0.f, 0.f};
    for (int t2 = 0; t2 < s; ++t2) {
      const float4 rv =
          reinterpret_cast<const float4*>(refb + (size_t)t2 * DN)[d4];
      const float w = sm.wb[t2];
      acc.x = fmaf(w, rv.x, acc.x); acc.y = fmaf(w, rv.y, acc.y);
      acc.z = fmaf(w, rv.z, acc.z); acc.w = fmaf(w, rv.w, acc.w);
    }
    const float4 r4 = {q4.x + acc.x, q4.y + acc.y, q4.z + acc.z, q4.w + acc.w};
    reinterpret_cast<float4*>(rawb + (size_t)s * DN)[d4] = q4;
    reinterpret_cast<float4*>(refb + (size_t)s * DN)[d4] = r4;
    if (r_to_zb) {                                  // r0 -> conv1 input chans
      const int d = d4 * 4;
      const int row = d >> 8, ch = (d >> 5) & 7, c0 = d & 31;
      sm.zbp[row][c0 + 1][ch] = tobf(r4.x);
      sm.zbp[row][c0 + 2][ch] = tobf(r4.y);
      sm.zbp[row][c0 + 3][ch] = tobf(r4.z);
      sm.zbp[row][c0 + 4][ch] = tobf(r4.w);
    } else {
      reinterpret_cast<float4*>(r_out_global)[d4] = r4;
    }
  }
}

__global__ __launch_bounds__(NTHREADS) void clsa_encoder_kernel(
    const float* __restrict__ x, const float* __restrict__ W0,
    const float* __restrict__ b0, const float* __restrict__ W1,
    const float* __restrict__ b1, float* __restrict__ out,
    float* __restrict__ ws) {
  __shared__ SMem sm;
  const int tid  = threadIdx.x;
  const int wave = tid >> 5;
  const int lane = tid & 31;
  const int b    = blockIdx.x;
  const bool hi  = lane >= 16;

  const size_t HIST = (size_t)BN * TN * DN;
  float* raw0b = ws + (size_t)b * TN * DN;
  float* ref0b = ws + HIST + (size_t)b * TN * DN;
  float* raw1b = ws + 2 * HIST + (size_t)b * TN * DN;
  float* ref1b = ws + 3 * HIST + (size_t)b * TN * DN;
  float* c0g   = ws + 4 * HIST + (size_t)b * DN;
  float* c1g   = ws + 4 * HIST + (size_t)BN * DN + (size_t)b * DN;

  float* enc = out;
  const size_t FIN = (size_t)BN * TN * DN;
  float* attn0b = out + FIN + 4 * (size_t)BN * DN + (size_t)b * TN * TN;
  float* attn1b = attn0b + (size_t)BN * TN * TN;

  if (tid < 32) { sm.bias0[tid] = b0[tid]; sm.bias1[tid] = b1[tid]; }

  // Weight A-fragments in kappa' = k*16+ic ordering (matches B fragments).
  AFrag a0[2][2], a1[2][2];   // [mtile][kchunk]
#pragma unroll
  for (int mt = 0; mt < 2; ++mt) {
    const int oc = mt * 16 + (lane & 15);
#pragma unroll
    for (int q = 0; q < 2; ++q) {
#pragma unroll
      for (int e = 0; e < 16; ++e) {
        const int kkA = (hi ? 16 : 0) + e;
        const int kp  = 32 * q + kkA;
        const int k   = kp >> 4, ic = kp & 15;
        a0[mt][q].e[e] =
            (k < 3 && ic < 9) ? tobf(W0[oc * 27 + ic * 3 + k]) : tobf(0.0f);
        a1[mt][q].e[e] = (k < 3) ? tobf(W1[oc * 48 + ic * 3 + k]) : tobf(0.0f);
      }
    }
  }

  // zero h-states, c-states, zbp halos (cols 0 and 33 never rewritten)
  for (int i = tid; i < DN; i += NTHREADS) {
    sm.h0[i] = 0.f; sm.h1[i] = 0.f; c0g[i] = 0.f; c1g[i] = 0.f;
  }
  for (int i = tid; i < ROWSN * 16 * 2; i += NTHREADS) {
    const int ic = i & 15, row = (i >> 4) & 31, side = i >> 9;
    sm.zbp[row][side ? (COLSN + 1) : 0][ic] = tobf(0.0f);
  }
  const float* xb = x + (size_t)b * TN * ROWSN * COLSN;
#if HAVE_TDM
  const unsigned xlds = (unsigned)(uintptr_t)&sm.xstage[0];
  if (wave == 0) tdm_load_frame(xb, xlds);      // stage frame 0
#endif
  __syncthreads();

  for (int s = 0; s < TN; ++s) {
    const float* xt = xb + (size_t)s * ROWSN * COLSN;
#if HAVE_TDM
    if (wave == 0) __builtin_amdgcn_s_wait_tensorcnt(0);
    __syncthreads();                             // publish TDM'd frame
#endif
    // stage conv0 input: ic0 = x_t, ic1..8 = h0, ic9..15 = zero pad
    for (int i = tid; i < ROWSN * 16 * COLSN; i += NTHREADS) {
      const int ic = i & 15, col = (i >> 4) & 31, row = i >> 9;
#if HAVE_TDM
      const float xv = sm.xstage[row * COLSN + col];
#else
      const float xv = xt[row * COLSN + col];
#endif
      const float v = (ic == 0) ? xv
                      : (ic <= 8) ? sm.h0[(row * CHN + ic - 1) * COLSN + col]
                                  : 0.f;
      sm.zbp[row][col + 1][ic] = tobf(v);
    }
    if (s + 2 < TN && tid < 64)                  // L2 prefetch two frames out
      __builtin_prefetch(xt + 2 * ROWSN * COLSN + tid * 16, 0, 1);
    __syncthreads();
#if HAVE_TDM
    if (s + 1 < TN && wave == 0)
      tdm_load_frame(xb + (size_t)(s + 1) * ROWSN * COLSN, xlds);
#endif

    conv_cell(sm, a0[0], a0[1], sm.bias0, c0g, sm.h0, wave, lane);
    __syncthreads();

    attention(sm, sm.h0, raw0b, ref0b, attn0b + (size_t)s * TN, s, nullptr,
              true, tid, wave, lane);
    for (int d4 = tid; d4 < DN / 4; d4 += NTHREADS) {   // old h1 -> ic 8..15
      const float4 h4 = reinterpret_cast<const float4*>(sm.h1)[d4];
      const int d = d4 * 4;
      const int row = d >> 8, ch = (d >> 5) & 7, c0 = d & 31;
      sm.zbp[row][c0 + 1][8 + ch] = tobf(h4.x);
      sm.zbp[row][c0 + 2][8 + ch] = tobf(h4.y);
      sm.zbp[row][c0 + 3][8 + ch] = tobf(h4.z);
      sm.zbp[row][c0 + 4][8 + ch] = tobf(h4.w);
    }
    __syncthreads();

    conv_cell(sm, a1[0], a1[1], sm.bias1, c1g, sm.h1, wave, lane);
    __syncthreads();

    attention(sm, sm.h1, raw1b, ref1b, attn1b + (size_t)s * TN, s,
              enc + ((size_t)b * TN + s) * DN, false, tid, wave, lane);
    __syncthreads();
  }

  for (int i = tid; i < DN; i += NTHREADS) {
    out[FIN + (size_t)b * DN + i]                       = sm.h0[i];
    out[FIN + (size_t)BN * DN + (size_t)b * DN + i]     = c0g[i];
    out[FIN + 2 * (size_t)BN * DN + (size_t)b * DN + i] = sm.h1[i];
    out[FIN + 3 * (size_t)BN * DN + (size_t)b * DN + i] = c1g[i];
  }
}

extern "C" void kernel_launch(void* const* d_in, const int* in_sizes, int n_in,
                              void* d_out, int out_size, void* d_ws,
                              size_t ws_size, hipStream_t stream) {
  (void)in_sizes; (void)n_in; (void)out_size; (void)ws_size;
  const float* x  = (const float*)d_in[0];
  const float* W0 = (const float*)d_in[1];
  const float* b0 = (const float*)d_in[2];
  const float* W1 = (const float*)d_in[3];
  const float* b1 = (const float*)d_in[4];
  clsa_encoder_kernel<<<BN, NTHREADS, 0, stream>>>(x, W0, b0, W1, b1,
                                                   (float*)d_out, (float*)d_ws);
}
